// _PointnetSAModuleBase_37237366456767
// MI455X (gfx1250) — matrix operand order repaired and checked by reading, not compiled, not run on hardware
//
#include <hip/hip_runtime.h>
#include <hip/hip_bf16.h>

typedef __attribute__((ext_vector_type(16))) _Float16 v16h;
typedef __attribute__((ext_vector_type(8)))  _Float16 v8h;
typedef __attribute__((ext_vector_type(8)))  float    v8f;

#define NB   8
#define NPT  8192
#define NM   1024
#define NK   32

// ---------------------------------------------------------------------------
// Kernel 1: Farthest point sampling. One block per batch, 1024 threads.
// Writes new_xyz directly into d_out[0 .. B*1024*3).
// ---------------------------------------------------------------------------
__global__ void __launch_bounds__(1024) fps_kernel(const float* __restrict__ xyz,
                                                   float* __restrict__ out_xyz) {
    int b = blockIdx.x;
    const float* P = xyz + (size_t)b * NPT * 3;
    __shared__ float dists[NPT];
    __shared__ float redv[32];
    __shared__ int   redi[32];
    __shared__ int   s_last;

    int t = threadIdx.x;
    int lane = t & 31;
    int wid  = t >> 5;

    for (int i = t; i < NPT; i += 1024) dists[i] = 1e10f;
    if (t == 0) {
        s_last = 0;
        out_xyz[(size_t)b * NM * 3 + 0] = P[0];
        out_xyz[(size_t)b * NM * 3 + 1] = P[1];
        out_xyz[(size_t)b * NM * 3 + 2] = P[2];
    }
    __syncthreads();

    int last = 0;
    for (int it = 1; it < NM; ++it) {
        float lx = P[last * 3 + 0];
        float ly = P[last * 3 + 1];
        float lz = P[last * 3 + 2];
        float bv = -1.0f;
        int   bi = 0;
        #pragma unroll
        for (int j = 0; j < 8; ++j) {
            int i = t + j * 1024;
            float dx = P[i * 3 + 0] - lx;
            float dy = P[i * 3 + 1] - ly;
            float dz = P[i * 3 + 2] - lz;
            // match JAX reduction order exactly: (dx*dx + dy*dy) + dz*dz, no fma
            float d = __fadd_rn(__fadd_rn(__fmul_rn(dx, dx), __fmul_rn(dy, dy)),
                                __fmul_rn(dz, dz));
            float dm = fminf(dists[i], d);
            dists[i] = dm;
            if (dm > bv) { bv = dm; bi = i; }   // ascending j -> first index wins
        }
        // wave-level argmax (first-index tie break)
        #pragma unroll
        for (int off = 16; off > 0; off >>= 1) {
            float ov = __shfl_down(bv, off, 32);
            int   oi = __shfl_down(bi, off, 32);
            if (ov > bv || (ov == bv && oi < bi)) { bv = ov; bi = oi; }
        }
        if (lane == 0) { redv[wid] = bv; redi[wid] = bi; }
        __syncthreads();
        if (wid == 0) {
            bv = redv[lane];
            bi = redi[lane];
            #pragma unroll
            for (int off = 16; off > 0; off >>= 1) {
                float ov = __shfl_down(bv, off, 32);
                int   oi = __shfl_down(bi, off, 32);
                if (ov > bv || (ov == bv && oi < bi)) { bv = ov; bi = oi; }
            }
            if (lane == 0) {
                s_last = bi;
                out_xyz[(size_t)b * NM * 3 + it * 3 + 0] = P[bi * 3 + 0];
                out_xyz[(size_t)b * NM * 3 + it * 3 + 1] = P[bi * 3 + 1];
                out_xyz[(size_t)b * NM * 3 + it * 3 + 2] = P[bi * 3 + 2];
            }
        }
        __syncthreads();
        last = s_last;
    }
}

// ---------------------------------------------------------------------------
// Kernel 2: pack weights into per-WMMA-fragment B layout (f16) in d_ws.
// Dense B (32x16, 8 VGPRs), contiguous k order used here: lane 0-15 -> N=lane, K=e (e=0..15);
//                           lane 16-31 -> N=lane-16, K=16+e.
// Fragment blob: frag*512 + lane*16 + e  (contiguous 32B per lane).
// ---------------------------------------------------------------------------
__global__ void wpack_kernel(const float* __restrict__ w1,
                             const float* __restrict__ w2,
                             const float* __restrict__ w3,
                             _Float16* __restrict__ wp) {
    int t = blockIdx.x * blockDim.x + threadIdx.x;
    int stride = gridDim.x * blockDim.x;
    // layer 1: 3 k-steps x 4 n-tiles, cin=67 padded to 96
    for (int e = t; e < 6144; e += stride) {
        int frag = e >> 9, within = e & 511;
        int L = within >> 4, el = within & 15;
        int ks = frag >> 2, nt = frag & 3;
        int K = el + ((L & 16) ? 16 : 0);
        int c = ks * 32 + K;
        int o = nt * 16 + (L & 15);
        wp[e] = (c < 67) ? (_Float16)w1[o * 67 + c] : (_Float16)0.f;
    }
    // layer 2: 2 k-steps x 4 n-tiles
    for (int e = t; e < 4096; e += stride) {
        int frag = e >> 9, within = e & 511;
        int L = within >> 4, el = within & 15;
        int ks = frag >> 2, nt = frag & 3;
        int K = el + ((L & 16) ? 16 : 0);
        int c = ks * 32 + K;
        int o = nt * 16 + (L & 15);
        wp[6144 + e] = (_Float16)w2[o * 64 + c];
    }
    // layer 3: 2 k-steps x 8 n-tiles
    for (int e = t; e < 8192; e += stride) {
        int frag = e >> 9, within = e & 511;
        int L = within >> 4, el = within & 15;
        int ks = frag >> 3, nt = frag & 7;
        int K = el + ((L & 16) ? 16 : 0);
        int c = ks * 32 + K;
        int o = nt * 16 + (L & 15);
        wp[10240 + e] = (_Float16)w3[o * 64 + c];
    }
}

// ---------------------------------------------------------------------------
// Kernel 3: ball query, one wave32 per query. First-32-in-radius in ascending
// index order (CUDA ball_query semantics), padded with the first hit.
// ---------------------------------------------------------------------------
__global__ void __launch_bounds__(256) ballq_kernel(const float* __restrict__ xyz,
                                                    const float* __restrict__ newxyz,
                                                    int* __restrict__ gidx) {
    __shared__ int slots[8][NK];
    int w    = threadIdx.x >> 5;
    int lane = threadIdx.x & 31;
    int q = blockIdx.x * 8 + w;
    int b = q >> 10;
    int m = q & 1023;
    const float* P = xyz + (size_t)b * NPT * 3;
    const float* C = newxyz + (size_t)b * NM * 3 + m * 3;
    float cx = C[0], cy = C[1], cz = C[2];
    const float r2 = (float)(0.4 * 0.4);

    int cnt = 0, first = 0;
    for (int base = 0; base < NPT && cnt < NK; base += 32) {
        int i = base + lane;
        float dx = P[i * 3 + 0] - cx;
        float dy = P[i * 3 + 1] - cy;
        float dz = P[i * 3 + 2] - cz;
        float d2 = __fadd_rn(__fadd_rn(__fmul_rn(dx, dx), __fmul_rn(dy, dy)),
                             __fmul_rn(dz, dz));
        bool pred = d2 < r2;
        unsigned mask = (unsigned)__ballot(pred);
        int rank = __popc(mask & ((1u << lane) - 1u));
        int slot = cnt + rank;
        if (pred && slot < NK) slots[w][slot] = i;
        if (cnt == 0 && mask) first = base + __ffs((int)mask) - 1;
        cnt += __popc(mask);
    }
    __syncthreads();
    int c = cnt < NK ? cnt : NK;
    int v = (lane < c) ? slots[w][lane] : first;
    gidx[(size_t)q * NK + lane] = v;
}

// ---------------------------------------------------------------------------
// Kernel 4: fused gather + 3-layer MLP (WMMA f16->f32) + max-pool.
// One wave per query (32 rows). Rows staged in LDS row-major f16 (stride 96),
// updated in place tile-by-tile across layers.
// ---------------------------------------------------------------------------
__device__ inline v16h load_a_frag(const _Float16* buf, int rowBase, int kbase,
                                   int lane) {
    // 16-bit A 16x32 fragment, contiguous k order (same as packed B):
    // lane 0-15:  M=lane,    K = kbase + {0..15}
    // lane 16-31: M=lane-16, K = kbase + {16..31}
    int r  = rowBase + (lane & 15);
    int kb = kbase + ((lane & 16) ? 16 : 0);
    const _Float16* p = buf + r * 96 + kb;
    v8h lo = *(const v8h*)p;
    v8h hi = *(const v8h*)(p + 8);
    v16h a;
    #pragma unroll
    for (int i = 0; i < 8; ++i) { a[i] = lo[i]; a[i + 8] = hi[i]; }
    return a;
}

#define WMMA(a, bm, c)                                                        \
    __builtin_amdgcn_wmma_f32_16x16x32_f16(false, (a), false, (bm), (short)0, \
                                           (c), false, false)

__global__ void __launch_bounds__(256) mlp_kernel(
    const float* __restrict__ xyz, const float* __restrict__ feats,
    const _Float16* __restrict__ wpack, const float* __restrict__ b1,
    const float* __restrict__ b2, const float* __restrict__ b3,
    const int* __restrict__ gidx, const float* __restrict__ newxyz,
    float* __restrict__ outf) {
    __shared__ __align__(16) _Float16 sbuf[8][32 * 96];  // 48 KB
    int w    = threadIdx.x >> 5;
    int lane = threadIdx.x & 31;
    int q = blockIdx.x * 8 + w;
    int b = q >> 10;
    int m = q & 1023;

    // ---- gather: row = this lane's neighbor, 67 channels (pad to 96) ----
    {
        int idx = gidx[(size_t)q * NK + lane];
        const float* C = newxyz + (size_t)b * NM * 3 + m * 3;
        const float* P = xyz + ((size_t)b * NPT + idx) * 3;
        _Float16* row = &sbuf[w][lane * 96];
        row[0] = (_Float16)(P[0] - C[0]);
        row[1] = (_Float16)(P[1] - C[1]);
        row[2] = (_Float16)(P[2] - C[2]);
        const float* F = feats + ((size_t)b * NPT + idx) * 64;
        #pragma unroll
        for (int c2 = 0; c2 < 64; ++c2) row[3 + c2] = (_Float16)F[c2];
        #pragma unroll
        for (int c2 = 67; c2 < 96; ++c2) row[c2] = (_Float16)0.f;
    }
    __syncthreads();

    const _Float16* sw  = &sbuf[w][0];
    _Float16*       swm = &sbuf[w][0];
    const _Float16* w1p = wpack;
    const _Float16* w2p = wpack + 6144;
    const _Float16* w3p = wpack + 10240;
    int ocol   = lane & 15;
    int rowoff = (lane & 16) ? 8 : 0;

    // ---- layer 1: 67(->96) -> 64, 3 k-steps x 4 n-tiles per row tile ----
    #pragma unroll
    for (int rt = 0; rt < 2; ++rt) {
        v8f acc[4] = {};
        #pragma unroll
        for (int ks = 0; ks < 3; ++ks) {
            v16h a = load_a_frag(sw, rt * 16, ks * 32, lane);
            #pragma unroll
            for (int nt = 0; nt < 4; ++nt) {
                v16h bf = *(const v16h*)(w1p + (ks * 4 + nt) * 512 + lane * 16);
                acc[nt] = WMMA(a, bf, acc[nt]);
            }
        }
        #pragma unroll
        for (int nt = 0; nt < 4; ++nt) {
            float bias = b1[nt * 16 + ocol];
            #pragma unroll
            for (int v = 0; v < 8; ++v) {
                float h = fmaxf(acc[nt][v] + bias, 0.f);
                swm[(rt * 16 + rowoff + v) * 96 + nt * 16 + ocol] = (_Float16)h;
            }
        }
    }
    __syncthreads();

    // ---- layer 2: 64 -> 64, in place ----
    #pragma unroll
    for (int rt = 0; rt < 2; ++rt) {
        v8f acc[4] = {};
        #pragma unroll
        for (int ks = 0; ks < 2; ++ks) {
            v16h a = load_a_frag(sw, rt * 16, ks * 32, lane);
            #pragma unroll
            for (int nt = 0; nt < 4; ++nt) {
                v16h bf = *(const v16h*)(w2p + (ks * 4 + nt) * 512 + lane * 16);
                acc[nt] = WMMA(a, bf, acc[nt]);
            }
        }
        #pragma unroll
        for (int nt = 0; nt < 4; ++nt) {
            float bias = b2[nt * 16 + ocol];
            #pragma unroll
            for (int v = 0; v < 8; ++v) {
                float h = fmaxf(acc[nt][v] + bias, 0.f);
                swm[(rt * 16 + rowoff + v) * 96 + nt * 16 + ocol] = (_Float16)h;
            }
        }
    }
    __syncthreads();

    // ---- layer 3: 64 -> 128, fused max-pool over the 32 rows ----
    float vmax[8];
    #pragma unroll
    for (int nt = 0; nt < 8; ++nt) vmax[nt] = -1e30f;
    #pragma unroll
    for (int rt = 0; rt < 2; ++rt) {
        v8f acc[8] = {};
        #pragma unroll
        for (int ks = 0; ks < 2; ++ks) {
            v16h a = load_a_frag(sw, rt * 16, ks * 32, lane);
            #pragma unroll
            for (int nt = 0; nt < 8; ++nt) {
                v16h bf = *(const v16h*)(w3p + (ks * 8 + nt) * 512 + lane * 16);
                acc[nt] = WMMA(a, bf, acc[nt]);
            }
        }
        #pragma unroll
        for (int nt = 0; nt < 8; ++nt)
            #pragma unroll
            for (int v = 0; v < 8; ++v) vmax[nt] = fmaxf(vmax[nt], acc[nt][v]);
    }
    // lanes L and L+16 hold the same output column (rows 0-7 vs 8-15)
    #pragma unroll
    for (int nt = 0; nt < 8; ++nt) {
        float mv = vmax[nt];
        mv = fmaxf(mv, __shfl_xor(mv, 16, 32));
        mv = fmaxf(mv + b3[nt * 16 + ocol], 0.f);  // relu(max+b) == max(relu(h+b))
        if (lane < 16)
            outf[(size_t)b * 128 * NM + (size_t)(nt * 16 + ocol) * NM + m] = mv;
    }
}

// ---------------------------------------------------------------------------
extern "C" void kernel_launch(void* const* d_in, const int* in_sizes, int n_in,
                              void* d_out, int out_size, void* d_ws, size_t ws_size,
                              hipStream_t stream) {
    (void)in_sizes; (void)n_in; (void)out_size; (void)ws_size;
    const float* xyz   = (const float*)d_in[0];
    const float* feats = (const float*)d_in[1];
    const float* w1    = (const float*)d_in[2];
    const float* b1    = (const float*)d_in[3];
    const float* w2    = (const float*)d_in[4];
    const float* b2    = (const float*)d_in[5];
    const float* w3    = (const float*)d_in[6];
    const float* b3    = (const float*)d_in[7];

    float* out    = (float*)d_out;
    float* newxyz = out;                       // B*1024*3 floats
    float* outf   = out + (size_t)NB * NM * 3; // B*128*1024 floats

    int*      gidx = (int*)d_ws;                                    // 1 MB
    _Float16* wp   = (_Float16*)((char*)d_ws + (size_t)NB * NM * NK * 4);

    fps_kernel<<<NB, 1024, 0, stream>>>(xyz, newxyz);
    wpack_kernel<<<64, 256, 0, stream>>>(w1, w2, w3, wp);
    ballq_kernel<<<(NB * NM) / 8, 256, 0, stream>>>(xyz, newxyz, gidx);
    mlp_kernel<<<(NB * NM) / 8, 256, 0, stream>>>(xyz, feats, wp, b1, b2, b3,
                                                  gidx, newxyz, outf);
}
